// FastformerBlock_7138235646419
// MI455X (gfx1250) — compile-verified, matrix operand order checked by reading, not run on hardware
//
#include <hip/hip_runtime.h>
#include <hip/hip_bf16.h>
#include <math.h>

typedef __bf16 bf16;
typedef __attribute__((ext_vector_type(16))) __bf16 v16bf;
typedef __attribute__((ext_vector_type(8)))  __bf16 v8bf;
typedef __attribute__((ext_vector_type(8)))  float   v8f;
typedef __attribute__((ext_vector_type(4)))  float   v4f;
typedef __attribute__((ext_vector_type(4)))  int     v4i;

#define BT    16384   // B*T
#define CDIM  1024
#define N1    3072    // 3*C
#define TSEQ  4096
#define NHEAD 16
#define HDIM  64

#define TILE_M 128    // block tile M
#define TILE_N 256    // block tile N
#define TILE_K 32
#define LDSS   40     // padded bf16 row stride (32 + 8), 80B = 5*16B aligned

#if defined(__has_builtin)
#if __has_builtin(__builtin_amdgcn_global_load_async_to_lds_b128) && \
    __has_builtin(__builtin_amdgcn_s_wait_asynccnt)
#define USE_ASYNC_LDS 1
#endif
#endif

#ifdef USE_ASYNC_LDS
typedef __attribute__((address_space(1))) v4i gv4i;   // global 128-bit chunk
typedef __attribute__((address_space(3))) v4i lv4i;   // LDS 128-bit chunk
#endif

union FragBF { v16bf v; v8bf h[2]; };

// ---------------------------------------------------------------------------
// Prep: W_qkv [1024][3072] fp32 -> Wt [3072][1024] bf16 (tiled transpose)
// ---------------------------------------------------------------------------
__global__ __launch_bounds__(256) void transpose_wqkv_kernel(
    const float* __restrict__ w, bf16* __restrict__ wt) {
  __shared__ bf16 ts[32][33];
  const int kb = blockIdx.x * 32;
  const int nb = blockIdx.y * 32;
  const int tx = threadIdx.x & 31;
  const int ty = threadIdx.x >> 5;
#pragma unroll
  for (int r = ty; r < 32; r += 8)
    ts[r][tx] = (bf16)w[(size_t)(kb + r) * N1 + nb + tx];
  __syncthreads();
#pragma unroll
  for (int r = ty; r < 32; r += 8)
    wt[(size_t)(nb + r) * CDIM + kb + tx] = ts[tx][r];
}

// ---------------------------------------------------------------------------
// Prep: W_v [1024][1024] fp32 -> bf16 (already [n][k] for u @ Wv^T)
// ---------------------------------------------------------------------------
__global__ __launch_bounds__(256) void convert_wv_kernel(
    const float* __restrict__ w, bf16* __restrict__ wb) {
  const int i = blockIdx.x * 256 + threadIdx.x;
  wb[i] = (bf16)w[i];
}

// ---------------------------------------------------------------------------
// Shared GEMM helpers
// ---------------------------------------------------------------------------
__device__ __forceinline__ void cvt16_to_bf16(const float* xp, v8bf& pa,
                                              v8bf& pb) {
  v4f xa = *(const v4f*)(xp + 0);
  v4f xb = *(const v4f*)(xp + 4);
  v4f xc = *(const v4f*)(xp + 8);
  v4f xd = *(const v4f*)(xp + 12);
#pragma unroll
  for (int j = 0; j < 4; ++j) {
    pa[j]     = (bf16)xa[j];
    pa[j + 4] = (bf16)xb[j];
    pb[j]     = (bf16)xc[j];
    pb[j + 4] = (bf16)xd[j];
  }
}

__device__ __forceinline__ void b_tile_fill(const bf16* gp, bf16* lp
#ifndef USE_ASYNC_LDS
                                            , v8bf* breg, bool do_store
#endif
) {
#ifdef USE_ASYNC_LDS
  // 4 x 16B async copies; IOFFSET advances both sides identically (ISA 10.x)
  gv4i* g = (gv4i*)gp;
  lv4i* l = (lv4i*)lp;
  __builtin_amdgcn_global_load_async_to_lds_b128(g, l, 0, 0);
  __builtin_amdgcn_global_load_async_to_lds_b128(g, l, 16, 0);
  __builtin_amdgcn_global_load_async_to_lds_b128(g, l, 32, 0);
  __builtin_amdgcn_global_load_async_to_lds_b128(g, l, 48, 0);
#else
  if (do_store) {
#pragma unroll
    for (int c = 0; c < 4; ++c) *(v8bf*)(lp + c * 8) = breg[c];
  }
#pragma unroll
  for (int c = 0; c < 4; ++c) breg[c] = *(const v8bf*)(gp + c * 8);
#endif
}

// Compute one 64x64 wave tile from LDS buffers: all 8 fragment loads issued
// up-front so the scheduler can use staggered s_wait_dscnt partial waits.
__device__ __forceinline__ void compute_tile(const bf16* Asb, const bf16* Bsb,
                                             int wm, int wn, int lrow, int hi,
                                             v8f (&acc)[4][4]) {
  FragBF afrag[4], bfrag[4];
#pragma unroll
  for (int i = 0; i < 4; ++i) {
    const bf16* base = Asb + (wm + i * 16 + lrow) * LDSS + hi * 8;
    afrag[i].h[0] = *(const v8bf*)base;
    afrag[i].h[1] = *(const v8bf*)(base + 16);
  }
#pragma unroll
  for (int j = 0; j < 4; ++j) {
    const bf16* base = Bsb + (wn + j * 16 + lrow) * LDSS + hi * 8;
    bfrag[j].h[0] = *(const v8bf*)base;
    bfrag[j].h[1] = *(const v8bf*)(base + 16);
  }
#pragma unroll
  for (int j = 0; j < 4; ++j)
#pragma unroll
    for (int i = 0; i < 4; ++i)
      acc[i][j] = __builtin_amdgcn_wmma_f32_16x16x32_bf16(
          false, afrag[i].v, false, bfrag[j].v, (short)0, acc[i][j], false,
          false);
}

// ---------------------------------------------------------------------------
// GEMM1: qkv[16384][3072] (bf16) = x (f32) @ Wt^T   (Wt is [n][k] bf16)
// block tile 128x256, 8 waves (2m x 4n), wave tile 64x64, double-buffered LDS
// ---------------------------------------------------------------------------
__global__ __launch_bounds__(256) void gemm_qkv_kernel(
    const float* __restrict__ x, const bf16* __restrict__ wt,
    bf16* __restrict__ qkv) {
  __shared__ bf16 As[2][TILE_M * LDSS];   // 20 KB
  __shared__ bf16 Bs[2][TILE_N * LDSS];   // 40 KB

  const int tid  = threadIdx.x;
  const int m0   = blockIdx.x * TILE_M;
  const int n0   = blockIdx.y * TILE_N;
  const int lane = tid & 31;
  const int wid  = tid >> 5;
  const int wm   = (wid & 1) * 64;
  const int wn   = (wid >> 1) * 64;
  const int lrow = lane & 15;
  const int hi   = lane >> 4;

  const int arow = tid >> 1;          // A coop row 0..127
  const int akh  = (tid & 1) * 16;    // A k half
  const int brow = tid;               // B coop row 0..255

  const float* xrow = x  + (size_t)(m0 + arow) * CDIM + akh;
  const bf16*  wrow = wt + (size_t)(n0 + brow) * CDIM;

  v8f acc[4][4] = {};
  v8bf pa, pb;                        // staged A (converted)
#ifndef USE_ASYNC_LDS
  v8bf breg[4];
#endif

  // ---- prologue: tile 0 ----
  cvt16_to_bf16(xrow, pa, pb);
#ifdef USE_ASYNC_LDS
  b_tile_fill(wrow, &Bs[0][brow * LDSS]);
#else
  b_tile_fill(wrow, nullptr, breg, false);
#endif

  int buf = 0;
  for (int k0 = 0; k0 < CDIM - TILE_K; k0 += TILE_K) {
    *(v8bf*)&As[buf][arow * LDSS + akh]     = pa;
    *(v8bf*)&As[buf][arow * LDSS + akh + 8] = pb;
#ifdef USE_ASYNC_LDS
    __builtin_amdgcn_s_wait_asynccnt(0);   // B(tile k0) resident
#else
#pragma unroll
    for (int c = 0; c < 4; ++c) *(v8bf*)&Bs[buf][brow * LDSS + c * 8] = breg[c];
#endif
    __syncthreads();

    // unconditional next-tile fills (target buffer's readers are done)
    const int kn = k0 + TILE_K;
#ifdef USE_ASYNC_LDS
    b_tile_fill(wrow + kn, &Bs[buf ^ 1][brow * LDSS]);
#else
    b_tile_fill(wrow + kn, nullptr, breg, false);
#endif
    cvt16_to_bf16(xrow + kn, pa, pb);

    compute_tile(&As[buf][0], &Bs[buf][0], wm, wn, lrow, hi, acc);
    buf ^= 1;
  }
  // ---- final tile (no prefetch) ----
  *(v8bf*)&As[buf][arow * LDSS + akh]     = pa;
  *(v8bf*)&As[buf][arow * LDSS + akh + 8] = pb;
#ifdef USE_ASYNC_LDS
  __builtin_amdgcn_s_wait_asynccnt(0);
#else
#pragma unroll
  for (int c = 0; c < 4; ++c) *(v8bf*)&Bs[buf][brow * LDSS + c * 8] = breg[c];
#endif
  __syncthreads();
  compute_tile(&As[buf][0], &Bs[buf][0], wm, wn, lrow, hi, acc);

  // ---- epilogue ----
#pragma unroll
  for (int i = 0; i < 4; ++i)
#pragma unroll
    for (int j = 0; j < 4; ++j)
#pragma unroll
      for (int r = 0; r < 8; ++r) {
        const int m = m0 + wm + i * 16 + r + hi * 8;
        const int n = n0 + wn + j * 16 + lrow;
        qkv[(size_t)m * N1 + n] = (bf16)acc[i][j][r];
      }
}

// ---------------------------------------------------------------------------
// Pool q: per (b,h) block. alpha=softmax_T(q.w_q+b_q); q_att=sum alpha*q
// ---------------------------------------------------------------------------
__global__ __launch_bounds__(256) void pool_q_kernel(
    const bf16* __restrict__ qkv, const float* __restrict__ w_q,
    const float* __restrict__ b_q, float* __restrict__ q_att) {
  __shared__ float logits[TSEQ];
  __shared__ float wq_s[HDIM];
  __shared__ float red[256];
  const int b = blockIdx.x >> 4, h = blockIdx.x & 15;
  const int tid = threadIdx.x;
  if (tid < HDIM) wq_s[tid] = w_q[tid];
  __syncthreads();

  const bf16* qbase = qkv + (size_t)b * TSEQ * N1 + h * HDIM;
  const float bias = b_q[0];
  float lmax = -INFINITY;
  for (int t = tid; t < TSEQ; t += 256) {
    const bf16* qp = qbase + (size_t)t * N1;
    float s = bias;
#pragma unroll
    for (int d = 0; d < HDIM; ++d) s += (float)qp[d] * wq_s[d];
    logits[t] = s;
    lmax = fmaxf(lmax, s);
  }
  red[tid] = lmax; __syncthreads();
  for (int s2 = 128; s2 > 0; s2 >>= 1) {
    if (tid < s2) red[tid] = fmaxf(red[tid], red[tid + s2]);
    __syncthreads();
  }
  lmax = red[0]; __syncthreads();

  float lsum = 0.f;
  for (int t = tid; t < TSEQ; t += 256) {
    const float e = __expf(logits[t] - lmax);
    logits[t] = e;
    lsum += e;
  }
  red[tid] = lsum; __syncthreads();
  for (int s2 = 128; s2 > 0; s2 >>= 1) {
    if (tid < s2) red[tid] += red[tid + s2];
    __syncthreads();
  }
  const float inv = 1.f / red[0];
  __syncthreads();

  const int d = tid & 63, tg = tid >> 6;
  float acc = 0.f;
  for (int t = tg; t < TSEQ; t += 4)
    acc += logits[t] * (float)qbase[(size_t)t * N1 + d];
  red[tid] = acc; __syncthreads();
  if (tg == 0)
    q_att[(b * NHEAD + h) * HDIM + d] =
        (red[d] + red[64 + d] + red[128 + d] + red[192 + d]) * inv;
}

// ---------------------------------------------------------------------------
// Pool k: beta=softmax_T((q_att*k).w_k+b_k); k_att = q_att * sum beta*k
// ---------------------------------------------------------------------------
__global__ __launch_bounds__(256) void pool_k_kernel(
    const bf16* __restrict__ qkv, const float* __restrict__ w_k,
    const float* __restrict__ b_k, const float* __restrict__ q_att,
    float* __restrict__ k_att) {
  __shared__ float logits[TSEQ];
  __shared__ float ew_s[HDIM];
  __shared__ float qa_s[HDIM];
  __shared__ float red[256];
  const int b = blockIdx.x >> 4, h = blockIdx.x & 15;
  const int tid = threadIdx.x;
  if (tid < HDIM) {
    const float qa = q_att[(b * NHEAD + h) * HDIM + tid];
    qa_s[tid] = qa;
    ew_s[tid] = qa * w_k[tid];
  }
  __syncthreads();

  const bf16* kbase = qkv + (size_t)b * TSEQ * N1 + CDIM + h * HDIM;
  const float bias = b_k[0];
  float lmax = -INFINITY;
  for (int t = tid; t < TSEQ; t += 256) {
    const bf16* kp = kbase + (size_t)t * N1;
    float s = bias;
#pragma unroll
    for (int d = 0; d < HDIM; ++d) s += (float)kp[d] * ew_s[d];
    logits[t] = s;
    lmax = fmaxf(lmax, s);
  }
  red[tid] = lmax; __syncthreads();
  for (int s2 = 128; s2 > 0; s2 >>= 1) {
    if (tid < s2) red[tid] = fmaxf(red[tid], red[tid + s2]);
    __syncthreads();
  }
  lmax = red[0]; __syncthreads();

  float lsum = 0.f;
  for (int t = tid; t < TSEQ; t += 256) {
    const float e = __expf(logits[t] - lmax);
    logits[t] = e;
    lsum += e;
  }
  red[tid] = lsum; __syncthreads();
  for (int s2 = 128; s2 > 0; s2 >>= 1) {
    if (tid < s2) red[tid] += red[tid + s2];
    __syncthreads();
  }
  const float inv = 1.f / red[0];
  __syncthreads();

  const int d = tid & 63, tg = tid >> 6;
  float acc = 0.f;
  for (int t = tg; t < TSEQ; t += 4)
    acc += logits[t] * (float)kbase[(size_t)t * N1 + d];
  red[tid] = acc; __syncthreads();
  if (tg == 0)
    k_att[(b * NHEAD + h) * HDIM + d] =
        qa_s[d] * (red[d] + red[64 + d] + red[128 + d] + red[192 + d]) * inv;
}

// ---------------------------------------------------------------------------
// GEMM2: out[16384][1024] = (k_att*v) @ Wv^T + b_v + q   (fp32 out)
// ---------------------------------------------------------------------------
__global__ __launch_bounds__(256) void gemm_out_kernel(
    const bf16* __restrict__ qkv, const bf16* __restrict__ wv,
    const float* __restrict__ katt, const float* __restrict__ bv,
    float* __restrict__ out) {
  __shared__ bf16 As[2][TILE_M * LDSS];
  __shared__ bf16 Bs[2][TILE_N * LDSS];

  const int tid  = threadIdx.x;
  const int m0   = blockIdx.x * TILE_M;
  const int n0   = blockIdx.y * TILE_N;
  const int bsel = m0 >> 12;
  const int lane = tid & 31;
  const int wid  = tid >> 5;
  const int wm   = (wid & 1) * 64;
  const int wn   = (wid >> 1) * 64;
  const int lrow = lane & 15;
  const int hi   = lane >> 4;

  const int arow = tid >> 1;
  const int akh  = (tid & 1) * 16;
  const int brow = tid;

  v8f acc[4][4] = {};
  v8bf pa, pb;
#ifndef USE_ASYNC_LDS
  v8bf breg[4];
#endif

  const bf16*  vrow = qkv + (size_t)(m0 + arow) * N1 + 2 * CDIM + akh;
  const float* krow = katt + bsel * CDIM + akh;
  const bf16*  wrow = wv + (size_t)(n0 + brow) * CDIM;

  // staged A build: u = k_att[b][k] * v[m][k]
  {
    v8bf v0 = *(const v8bf*)vrow;
    v8bf v1 = *(const v8bf*)(vrow + 8);
#pragma unroll
    for (int j = 0; j < 8; ++j) {
      pa[j] = (bf16)((float)v0[j] * krow[j]);
      pb[j] = (bf16)((float)v1[j] * krow[8 + j]);
    }
  }
#ifdef USE_ASYNC_LDS
  b_tile_fill(wrow, &Bs[0][brow * LDSS]);
#else
  b_tile_fill(wrow, nullptr, breg, false);
#endif

  int buf = 0;
  for (int k0 = 0; k0 < CDIM - TILE_K; k0 += TILE_K) {
    *(v8bf*)&As[buf][arow * LDSS + akh]     = pa;
    *(v8bf*)&As[buf][arow * LDSS + akh + 8] = pb;
#ifdef USE_ASYNC_LDS
    __builtin_amdgcn_s_wait_asynccnt(0);
#else
#pragma unroll
    for (int c = 0; c < 4; ++c) *(v8bf*)&Bs[buf][brow * LDSS + c * 8] = breg[c];
#endif
    __syncthreads();

    const int kn = k0 + TILE_K;
#ifdef USE_ASYNC_LDS
    b_tile_fill(wrow + kn, &Bs[buf ^ 1][brow * LDSS]);
#else
    b_tile_fill(wrow + kn, nullptr, breg, false);
#endif
    {
      v8bf v0 = *(const v8bf*)(vrow + kn);
      v8bf v1 = *(const v8bf*)(vrow + kn + 8);
#pragma unroll
      for (int j = 0; j < 8; ++j) {
        pa[j] = (bf16)((float)v0[j] * krow[kn + j]);
        pb[j] = (bf16)((float)v1[j] * krow[kn + 8 + j]);
      }
    }

    compute_tile(&As[buf][0], &Bs[buf][0], wm, wn, lrow, hi, acc);
    buf ^= 1;
  }
  // ---- final tile ----
  *(v8bf*)&As[buf][arow * LDSS + akh]     = pa;
  *(v8bf*)&As[buf][arow * LDSS + akh + 8] = pb;
#ifdef USE_ASYNC_LDS
  __builtin_amdgcn_s_wait_asynccnt(0);
#else
#pragma unroll
  for (int c = 0; c < 4; ++c) *(v8bf*)&Bs[buf][brow * LDSS + c * 8] = breg[c];
#endif
  __syncthreads();
  compute_tile(&As[buf][0], &Bs[buf][0], wm, wn, lrow, hi, acc);

  // ---- epilogue: + b_v[n] + q[m][n] ----
#pragma unroll
  for (int i = 0; i < 4; ++i)
#pragma unroll
    for (int j = 0; j < 4; ++j) {
      const int n = n0 + wn + j * 16 + lrow;
      const float bvn = bv[n];
#pragma unroll
      for (int r = 0; r < 8; ++r) {
        const int m = m0 + wm + i * 16 + r + hi * 8;
        const float qv = (float)qkv[(size_t)m * N1 + n];
        out[(size_t)m * CDIM + n] = acc[i][j][r] + bvn + qv;
      }
    }
}

// ---------------------------------------------------------------------------
// Host launcher
// ---------------------------------------------------------------------------
extern "C" void kernel_launch(void* const* d_in, const int* in_sizes, int n_in,
                              void* d_out, int out_size, void* d_ws,
                              size_t ws_size, hipStream_t stream) {
  (void)in_sizes; (void)n_in; (void)out_size; (void)ws_size;
  const float* x     = (const float*)d_in[0];
  const float* W_qkv = (const float*)d_in[1];
  const float* w_q   = (const float*)d_in[2];
  const float* b_q   = (const float*)d_in[3];
  const float* w_k   = (const float*)d_in[4];
  const float* b_k   = (const float*)d_in[5];
  const float* W_v   = (const float*)d_in[6];
  const float* b_v   = (const float*)d_in[7];
  float* out = (float*)d_out;

  char* ws = (char*)d_ws;
  const size_t QKV_BYTES = (size_t)BT * N1 * sizeof(bf16);      // 96 MB
  const size_t WT_BYTES  = (size_t)N1 * CDIM * sizeof(bf16);    // 6 MB
  const size_t WV_BYTES  = (size_t)CDIM * CDIM * sizeof(bf16);  // 2 MB
  bf16*  qkv  = (bf16*)ws;
  bf16*  wt   = (bf16*)(ws + QKV_BYTES);
  bf16*  wv   = (bf16*)(ws + QKV_BYTES + WT_BYTES);
  float* qatt = (float*)(ws + QKV_BYTES + WT_BYTES + WV_BYTES);
  float* katt = qatt + 4 * CDIM;

  transpose_wqkv_kernel<<<dim3(CDIM / 32, N1 / 32), 256, 0, stream>>>(W_qkv, wt);
  convert_wv_kernel<<<(CDIM * CDIM) / 256, 256, 0, stream>>>(W_v, wv);
  gemm_qkv_kernel<<<dim3(BT / TILE_M, N1 / TILE_N), 256, 0, stream>>>(x, wt, qkv);
  pool_q_kernel<<<4 * NHEAD, 256, 0, stream>>>(qkv, w_q, b_q, qatt);
  pool_k_kernel<<<4 * NHEAD, 256, 0, stream>>>(qkv, w_k, b_k, qatt, katt);
  gemm_out_kernel<<<dim3(BT / TILE_M, CDIM / TILE_N), 256, 0, stream>>>(
      qkv, wv, katt, b_v, out);
}
